// MambaTwo_53231824666651
// MI455X (gfx1250) — compile-verified, matrix operand order checked by reading, not run on hardware
//
#include <hip/hip_runtime.h>
#include <stdint.h>

// ---------------- model constants ----------------
#define B_SZ     2
#define L_SZ     1024
#define NTOK     (B_SZ * L_SZ)      // 2048
#define DMODEL   768
#define DINNER   1536
#define DSTATE   16
#define DTRANK   48
#define DTPAD    64
#define DCONV    4
#define VOCAB    32000
#define XPROJ_N  (DTRANK + 2 * DSTATE)   // 80
#define XPROJ_NP 256                      // padded to one workgroup tile

typedef __bf16 bf16;
typedef __attribute__((ext_vector_type(16))) __bf16   bf16x16;
typedef __attribute__((ext_vector_type(8)))  float    f32x8;
typedef __attribute__((ext_vector_type(4)))  uint32_t u32x4;

union Frag16 { bf16x16 v; u32x4 q[2]; };
union Frag8f { f32x8  v; float f[8]; };

// ---------------- WMMA GEMM: C[M,N] = A[M,K](bf16) * W[N,K]^T(bf16), f32 out ------
// Requirements (guaranteed by caller): M % 128 == 0 and grid.x == M/128;
// W physically padded to grid.y*256 rows; K % 32 == 0.
// wave32: each wave -> 64x64 tile (4x4 WMMA 16x16x32 bf16); 8 waves -> 128x256/WG.
__global__ __launch_bounds__(256)
void gemm_bf16_nt(const bf16* __restrict__ A, const bf16* __restrict__ W,
                  float* __restrict__ C, int M, int N, int K) {
    const int lane = threadIdx.x & 31;
    const int wave = threadIdx.x >> 5;
    const int m0 = blockIdx.x * 128 + (wave >> 2) * 64;
    const int n0 = blockIdx.y * 256 + (wave & 3) * 64;
    const int lm = lane & 15;
    const int kg = (lane >> 4) << 3;          // K sub-group: 0 or 8

    Frag8f acc[4][4] = {};
    const bf16* aBase = A + (size_t)(m0 + lm) * K + kg;   // per ISA 16-bit A layout
    const bf16* wBase = W + (size_t)(n0 + lm) * K + kg;   // B symmetric over N

    for (int kb = 0; kb < K; kb += 32) {
        Frag16 af[4], bfm[4];
        #pragma unroll
        for (int t = 0; t < 4; ++t) {
            const bf16* ap = aBase + (size_t)(t * 16) * K + kb;
            af[t].q[0]  = *(const u32x4*)(ap);        // K = kg+0..7   (as bf16 pairs)
            af[t].q[1]  = *(const u32x4*)(ap + 16);   // K = kg+16..23
            const bf16* wp = wBase + (size_t)(t * 16) * K + kb;
            bfm[t].q[0] = *(const u32x4*)(wp);
            bfm[t].q[1] = *(const u32x4*)(wp + 16);
        }
        if (kb + 64 <= K) {
            __builtin_prefetch(aBase + kb + 32, 0, 0);   // global_prefetch_b8
            __builtin_prefetch(wBase + kb + 32, 0, 0);
        }
        #pragma unroll
        for (int tr = 0; tr < 4; ++tr)
            #pragma unroll
            for (int tc = 0; tc < 4; ++tc)
                acc[tr][tc].v = __builtin_amdgcn_wmma_f32_16x16x32_bf16(
                    false, af[tr].v, false, bfm[tc].v,
                    (short)0, acc[tr][tc].v, false, false);
    }

    // C/D layout: VGPR r -> M = r + 8*(lane>=16), N = lane&15
    const int rbase = (lane < 16) ? 0 : 8;
    #pragma unroll
    for (int tr = 0; tr < 4; ++tr) {
        #pragma unroll
        for (int tc = 0; tc < 4; ++tc) {
            int n = n0 + tc * 16 + lm;
            if (n >= N) continue;                      // only x_proj (N=80) hits this
            float* cp = C + (size_t)(m0 + tr * 16 + rbase) * N + n;
            #pragma unroll
            for (int r = 0; r < 8; ++r)
                cp[(size_t)r * N] = acc[tr][tc].f[r];
        }
    }
}

// ---------------- elementwise / small kernels ----------------
__global__ void embed_kernel(const int* __restrict__ x, const float* __restrict__ E,
                             float* __restrict__ h) {
    size_t i = (size_t)blockIdx.x * blockDim.x + threadIdx.x;
    if (i >= (size_t)NTOK * DMODEL) return;
    int m = (int)(i / DMODEL), d = (int)(i % DMODEL);
    h[i] = E[(size_t)x[m] * DMODEL + d];
}

__global__ void cvt_kernel(const float* __restrict__ s, bf16* __restrict__ d, size_t n) {
    size_t i = (size_t)blockIdx.x * blockDim.x + threadIdx.x;
    if (i < n) d[i] = (bf16)s[i];
}

// dst[r,k] (r<dstRows, k<dstK) = (r<srcRows && k<copyK) ? src[r*srcStride+k] : 0
__global__ void pad2_cvt_kernel(const float* __restrict__ s, bf16* __restrict__ d,
                                int srcRows, int dstRows, int srcStride,
                                int copyK, int dstK) {
    size_t i = (size_t)blockIdx.x * blockDim.x + threadIdx.x;
    if (i >= (size_t)dstRows * dstK) return;
    int r = (int)(i / dstK), k = (int)(i % dstK);
    d[i] = (r < srcRows && k < copyK) ? (bf16)s[(size_t)r * srcStride + k] : (bf16)0.0f;
}

__global__ __launch_bounds__(256)
void rmsnorm_kernel(const float* __restrict__ X, const float* __restrict__ w,
                    bf16* __restrict__ out, int D) {
    int row = blockIdx.x;
    const float* x = X + (size_t)row * D;
    float s = 0.f;
    for (int i = threadIdx.x; i < D; i += 256) { float v = x[i]; s += v * v; }
    __shared__ float red[256];
    red[threadIdx.x] = s; __syncthreads();
    for (int o = 128; o > 0; o >>= 1) {
        if (threadIdx.x < o) red[threadIdx.x] += red[threadIdx.x + o];
        __syncthreads();
    }
    float inv = rsqrtf(red[0] / (float)D + 1e-5f);
    for (int i = threadIdx.x; i < D; i += 256)
        out[(size_t)row * D + i] = (bf16)(x[i] * inv * w[i]);
}

// causal depthwise conv (D_CONV=4) + bias + SiLU over first half of xz
__global__ void conv_silu_kernel(const float* __restrict__ xz,
                                 const float* __restrict__ cw, const float* __restrict__ cb,
                                 float* __restrict__ xa, bf16* __restrict__ xaBf) {
    size_t i = (size_t)blockIdx.x * blockDim.x + threadIdx.x;
    if (i >= (size_t)NTOK * DINNER) return;
    int m = (int)(i / DINNER), e = (int)(i % DINNER);
    int b = m / L_SZ, l = m % L_SZ;
    float acc = cb[e];
    #pragma unroll
    for (int t = 0; t < DCONV; ++t) {
        int ll = l - (DCONV - 1) + t;
        float v = (ll >= 0) ? xz[((size_t)b * L_SZ + ll) * (2 * DINNER) + e] : 0.f;
        acc += cw[e * DCONV + t] * v;
    }
    float y = acc / (1.f + __expf(-acc));     // SiLU
    xa[i] = y;
    xaBf[i] = (bf16)y;
}

__global__ void softplus_kernel(float* __restrict__ d, const float* __restrict__ bias, size_t n) {
    size_t i = (size_t)blockIdx.x * blockDim.x + threadIdx.x;
    if (i >= n) return;
    float v = d[i] + bias[i % DINNER];
    d[i] = (v > 20.f) ? v : log1pf(__expf(v));
}

// sequential selective scan: one thread per (b, channel e), 16 states in regs
__global__ __launch_bounds__(256)
void scan_kernel(const float* __restrict__ delta, const float* __restrict__ xa,
                 const float* __restrict__ dbc, const float* __restrict__ A_log,
                 const float* __restrict__ Dp, float* __restrict__ y) {
    int e = blockIdx.x * blockDim.x + threadIdx.x;
    int b = blockIdx.y;
    if (e >= DINNER) return;
    float Ae[DSTATE], h[DSTATE];
    #pragma unroll
    for (int n = 0; n < DSTATE; ++n) { Ae[n] = -__expf(A_log[e * DSTATE + n]); h[n] = 0.f; }
    float de = Dp[e];
    for (int l = 0; l < L_SZ; ++l) {
        size_t m = (size_t)b * L_SZ + l;
        float dt = delta[m * DINNER + e];
        float xv = xa[m * DINNER + e];
        const float* bc = dbc + m * XPROJ_N;
        float accv = 0.f;
        #pragma unroll
        for (int n = 0; n < DSTATE; ++n) {
            float dA  = __expf(dt * Ae[n]);
            float dBx = dt * bc[DTRANK + n] * xv;
            h[n] = dA * h[n] + dBx;
            accv += h[n] * bc[DTRANK + DSTATE + n];
        }
        y[m * DINNER + e] = accv + de * xv;
    }
}

__global__ void gate_kernel(const float* __restrict__ y, const float* __restrict__ xz,
                            bf16* __restrict__ yBf) {
    size_t i = (size_t)blockIdx.x * blockDim.x + threadIdx.x;
    if (i >= (size_t)NTOK * DINNER) return;
    int m = (int)(i / DINNER), e = (int)(i % DINNER);
    float z = xz[(size_t)m * (2 * DINNER) + DINNER + e];
    float sz = z / (1.f + __expf(-z));
    yBf[i] = (bf16)(y[i] * sz);
}

__global__ void add_res_kernel(float* __restrict__ h, const float* __restrict__ p, size_t n) {
    size_t i = (size_t)blockIdx.x * blockDim.x + threadIdx.x;
    if (i < n) h[i] += p[i];
}

__global__ void bias_kernel(float* __restrict__ o, const float* __restrict__ b, size_t n) {
    size_t i = (size_t)blockIdx.x * blockDim.x + threadIdx.x;
    if (i < n) o[i] += b[i % VOCAB];
}

// ---------------- workspace layout (bytes, all 256-aligned) ----------------
#define WS_HS      ((size_t)0)                                   // f32 [2048,768]
#define WS_XNBF    (WS_HS    + (size_t)NTOK * DMODEL * 4)        // bf16 [2048,768]
#define WS_XZ      (WS_XNBF  + (size_t)NTOK * DMODEL * 2)        // f32 [2048,3072]
#define WS_XA      (WS_XZ    + (size_t)NTOK * 2 * DINNER * 4)    // f32 [2048,1536]
#define WS_XABF    (WS_XA    + (size_t)NTOK * DINNER * 4)        // bf16
#define WS_DBC     (WS_XABF  + (size_t)NTOK * DINNER * 2)        // f32 [2048,80]
#define WS_DINBF   (WS_DBC   + (size_t)NTOK * XPROJ_N * 4)       // bf16 [2048,64]
#define WS_DELTA   (WS_DINBF + (size_t)NTOK * DTPAD * 2)         // f32 [2048,1536]
#define WS_Y       (WS_DELTA + (size_t)NTOK * DINNER * 4)        // f32 [2048,1536]
#define WS_YBF     (WS_Y     + (size_t)NTOK * DINNER * 4)        // bf16
#define WS_WIN     (WS_YBF   + (size_t)NTOK * DINNER * 2)        // bf16 [3072,768]
#define WS_WX      (WS_WIN   + (size_t)2 * DINNER * DMODEL * 2)  // bf16 [256,1536] (row-padded)
#define WS_WDT     (WS_WX    + (size_t)XPROJ_NP * DINNER * 2)    // bf16 [1536,64]
#define WS_WOUT    (WS_WDT   + (size_t)DINNER * DTPAD * 2)       // bf16 [768,1536]
// head weights (49 MB bf16) alias the layer-temporary region (dead at final stage)
#define WS_WHEAD   WS_XZ
// projOut (f32 [2048,768]) aliases delta (dead after scan)
#define WS_PROJ    WS_DELTA

static inline dim3 grid1(size_t n) { return dim3((unsigned)((n + 255) / 256)); }

extern "C" void kernel_launch(void* const* d_in, const int* in_sizes, int n_in,
                              void* d_out, int out_size, void* d_ws, size_t ws_size,
                              hipStream_t stream) {
    const int*   x         = (const int*)  d_in[0];
    const float* embed_W   = (const float*)d_in[1];
    const float* norm_w    = (const float*)d_in[2];
    const float* in_proj_w = (const float*)d_in[3];
    const float* conv_w    = (const float*)d_in[4];
    const float* conv_b    = (const float*)d_in[5];
    const float* x_proj_w  = (const float*)d_in[6];
    const float* dt_proj_w = (const float*)d_in[7];
    const float* dt_proj_b = (const float*)d_in[8];
    const float* A_log     = (const float*)d_in[9];
    const float* D_param   = (const float*)d_in[10];
    const float* out_pj_w  = (const float*)d_in[11];
    const float* normf_w   = (const float*)d_in[12];
    const float* head_w    = (const float*)d_in[13];
    const float* head_b    = (const float*)d_in[14];
    float* out = (float*)d_out;

    char* ws = (char*)d_ws;
    float* hs    = (float*)(ws + WS_HS);
    bf16*  xnBf  = (bf16*) (ws + WS_XNBF);
    float* xz    = (float*)(ws + WS_XZ);
    float* xa    = (float*)(ws + WS_XA);
    bf16*  xaBf  = (bf16*) (ws + WS_XABF);
    float* dbc   = (float*)(ws + WS_DBC);
    bf16*  dInBf = (bf16*) (ws + WS_DINBF);
    float* delta = (float*)(ws + WS_DELTA);
    float* yb    = (float*)(ws + WS_Y);
    bf16*  yBf   = (bf16*) (ws + WS_YBF);
    bf16*  wIn   = (bf16*) (ws + WS_WIN);
    bf16*  wX    = (bf16*) (ws + WS_WX);
    bf16*  wDt   = (bf16*) (ws + WS_WDT);
    bf16*  wOut  = (bf16*) (ws + WS_WOUT);
    bf16*  wHead = (bf16*) (ws + WS_WHEAD);
    float* projO = (float*)(ws + WS_PROJ);

    // embedding gather
    embed_kernel<<<grid1((size_t)NTOK * DMODEL), 256, 0, stream>>>(x, embed_W, hs);

    for (int i = 0; i < 2; ++i) {
        const float* inW  = in_proj_w + (size_t)i * 2 * DINNER * DMODEL;
        const float* cw   = conv_w    + (size_t)i * DINNER * DCONV;
        const float* cb   = conv_b    + (size_t)i * DINNER;
        const float* xpW  = x_proj_w  + (size_t)i * XPROJ_N * DINNER;
        const float* dtW  = dt_proj_w + (size_t)i * DINNER * DTRANK;
        const float* dtB  = dt_proj_b + (size_t)i * DINNER;
        const float* Al   = A_log     + (size_t)i * DINNER * DSTATE;
        const float* Dpp  = D_param   + (size_t)i * DINNER;
        const float* outW = out_pj_w  + (size_t)i * DMODEL * DINNER;
        const float* nw   = norm_w    + (size_t)i * DMODEL;

        // weight conversions to bf16 (wX row-padded 80->256; wDt K-padded 48->64)
        cvt_kernel<<<grid1((size_t)2 * DINNER * DMODEL), 256, 0, stream>>>(inW, wIn, (size_t)2 * DINNER * DMODEL);
        pad2_cvt_kernel<<<grid1((size_t)XPROJ_NP * DINNER), 256, 0, stream>>>(xpW, wX, XPROJ_N, XPROJ_NP, DINNER, DINNER, DINNER);
        pad2_cvt_kernel<<<grid1((size_t)DINNER * DTPAD), 256, 0, stream>>>(dtW, wDt, DINNER, DINNER, DTRANK, DTRANK, DTPAD);
        cvt_kernel<<<grid1((size_t)DMODEL * DINNER), 256, 0, stream>>>(outW, wOut, (size_t)DMODEL * DINNER);

        // rmsnorm -> bf16
        rmsnorm_kernel<<<NTOK, 256, 0, stream>>>(hs, nw, xnBf, DMODEL);
        // xz = xn @ inW^T  [2048,3072]
        gemm_bf16_nt<<<dim3(NTOK / 128, 2 * DINNER / 256), 256, 0, stream>>>(xnBf, wIn, xz, NTOK, 2 * DINNER, DMODEL);
        // depthwise conv + SiLU
        conv_silu_kernel<<<grid1((size_t)NTOK * DINNER), 256, 0, stream>>>(xz, cw, cb, xa, xaBf);
        // dbc = xa @ xpW^T  [2048,80]  (W padded to 256 rows; stores clipped to N=80)
        gemm_bf16_nt<<<dim3(NTOK / 128, 1), 256, 0, stream>>>(xaBf, wX, dbc, NTOK, XPROJ_N, DINNER);
        // delta-rank slice of dbc, K-padded 48->64
        pad2_cvt_kernel<<<grid1((size_t)NTOK * DTPAD), 256, 0, stream>>>(dbc, dInBf, NTOK, NTOK, XPROJ_N, DTRANK, DTPAD);
        // delta_raw = dIn @ dtW^T  [2048,1536]
        gemm_bf16_nt<<<dim3(NTOK / 128, DINNER / 256), 256, 0, stream>>>(dInBf, wDt, delta, NTOK, DINNER, DTPAD);
        // softplus(+bias) in place
        softplus_kernel<<<grid1((size_t)NTOK * DINNER), 256, 0, stream>>>(delta, dtB, (size_t)NTOK * DINNER);
        // selective scan
        scan_kernel<<<dim3(DINNER / 256, B_SZ), 256, 0, stream>>>(delta, xa, dbc, Al, Dpp, yb);
        // gate with SiLU(z) -> bf16
        gate_kernel<<<grid1((size_t)NTOK * DINNER), 256, 0, stream>>>(yb, xz, yBf);
        // projOut = y @ outW^T  [2048,768]   (aliases delta, dead by now)
        gemm_bf16_nt<<<dim3(NTOK / 128, DMODEL / 256), 256, 0, stream>>>(yBf, wOut, projO, NTOK, DMODEL, DINNER);
        // residual
        add_res_kernel<<<grid1((size_t)NTOK * DMODEL), 256, 0, stream>>>(hs, projO, (size_t)NTOK * DMODEL);
    }

    // final norm + head
    rmsnorm_kernel<<<NTOK, 256, 0, stream>>>(hs, normf_w, xnBf, DMODEL);
    cvt_kernel<<<grid1((size_t)VOCAB * DMODEL), 256, 0, stream>>>(head_w, wHead, (size_t)VOCAB * DMODEL);
    gemm_bf16_nt<<<dim3(NTOK / 128, VOCAB / 256), 256, 0, stream>>>(xnBf, wHead, out, NTOK, VOCAB, DMODEL);
    bias_kernel<<<grid1((size_t)NTOK * VOCAB), 256, 0, stream>>>(out, head_b, (size_t)NTOK * VOCAB);
}
